// LSTM_9972914061460
// MI455X (gfx1250) — matrix operand order repaired and checked by reading, not compiled, not run on hardware
//
#include <hip/hip_runtime.h>
#include <hip/hip_bf16.h>

typedef __attribute__((ext_vector_type(16))) _Float16 v16h;
typedef __attribute__((ext_vector_type(8)))  _Float16 v8h;
typedef __attribute__((ext_vector_type(8)))  float    v8f;
typedef __attribute__((ext_vector_type(4)))  int      v4i;

union AFrag { v16h v; v8h h[2]; };

__device__ __forceinline__ float sigmoidf_(float x) {
    return 1.0f / (1.0f + __expf(-x));
}

// ---------------------------------------------------------------------------
// CDNA5 async global->LDS copy (16B granularity), ASYNCcnt-tracked.
// Builtin signature (from hipcc diagnostic): arg0 = v4i* in AS1 (__device__),
// arg1 = v4i* in AS3 (__shared__), then imm offset + imm cpol.
// Falls back to a synchronous VGPR round-trip if the builtin is absent.
// ---------------------------------------------------------------------------
#if defined(__HIP_DEVICE_COMPILE__) && __has_builtin(__builtin_amdgcn_global_load_async_to_lds_b128)
#define ASYNC_LDS 1
__device__ __forceinline__ void async_copy16(const void* gsrc, void* ldst) {
    __attribute__((address_space(1))) v4i* g =
        (__attribute__((address_space(1))) v4i*)(uintptr_t)gsrc;
    __attribute__((address_space(3))) v4i* l =
        (__attribute__((address_space(3))) v4i*)(uintptr_t)ldst;
    __builtin_amdgcn_global_load_async_to_lds_b128(g, l, 0, 0);
}
__device__ __forceinline__ void async_wait0() {
#if __has_builtin(__builtin_amdgcn_s_wait_asynccnt)
    __builtin_amdgcn_s_wait_asynccnt(0);
#else
    asm volatile("s_wait_asynccnt 0x0" ::: "memory");
#endif
}
#else
#define ASYNC_LDS 0
__device__ __forceinline__ void async_copy16(const void* gsrc, void* ldst) {
    *(float4*)ldst = *(const float4*)gsrc;
}
__device__ __forceinline__ void async_wait0() {}
#endif

// ---------------------------------------------------------------------------
// f32 -> f16 conversion (weights), 4 elems/thread
// ---------------------------------------------------------------------------
__global__ void cvt_f16_kernel(const float* __restrict__ src,
                               _Float16* __restrict__ dst, int n) {
    int i = (blockIdx.x * blockDim.x + threadIdx.x) * 4;
    if (i + 3 < n) {
        float4 f = *(const float4*)(src + i);
        dst[i + 0] = (_Float16)f.x;
        dst[i + 1] = (_Float16)f.y;
        dst[i + 2] = (_Float16)f.z;
        dst[i + 3] = (_Float16)f.w;
    } else {
        for (int k = i; k < n; ++k) dst[k] = (_Float16)src[k];
    }
}

// bsum = bih + bhh
__global__ void bias_sum_kernel(const float* __restrict__ a,
                                const float* __restrict__ b,
                                float* __restrict__ o, int n) {
    int i = blockIdx.x * blockDim.x + threadIdx.x;
    if (i < n) o[i] = a[i] + b[i];
}

// ---------------------------------------------------------------------------
// Embedding gather + f32->f16, 8 elems/thread (row-contiguous)
// ---------------------------------------------------------------------------
__global__ void embed_f16_kernel(const int* __restrict__ x,
                                 const float* __restrict__ emb,
                                 _Float16* __restrict__ xe,
                                 long total_rows, int E) {
    long i = (long)blockIdx.x * blockDim.x + threadIdx.x;
    long base = i * 8;
    if (base >= total_rows * (long)E) return;
    int row = (int)(base / E);
    int e   = (int)(base % E);
    int idx = x[row];
    const float4* p = (const float4*)(emb + (size_t)idx * E + e);
    float4 f0 = p[0], f1 = p[1];
    v8h o;
    o[0] = (_Float16)f0.x; o[1] = (_Float16)f0.y;
    o[2] = (_Float16)f0.z; o[3] = (_Float16)f0.w;
    o[4] = (_Float16)f1.x; o[5] = (_Float16)f1.y;
    o[6] = (_Float16)f1.z; o[7] = (_Float16)f1.w;
    *(v8h*)(xe + base) = o;
}

// ---------------------------------------------------------------------------
// GEMM: D[M,N] = A[M,K](f16) @ Bw[N,K](f16)^T + bias[N], f32 accumulate/out.
// All 8 waves of a block share one 16-row A slab (tiles_n % 8 == 0), which is
// staged once into LDS via async-to-LDS DMA; A fragments then come from
// ds_load_b128 pairs while B streams from global (L2-resident weights).
// ---------------------------------------------------------------------------
template <int K>
__global__ __launch_bounds__(256)
void gemm_bias_f16_kernel(const _Float16* __restrict__ A,
                          const _Float16* __restrict__ Bw,
                          const float* __restrict__ bias,
                          float* __restrict__ D,
                          int M, int N) {
    __shared__ _Float16 atile[16 * K];   // 16 rows x K halfs (8 KB for K=256)

    const int tid  = threadIdx.x;
    const int wave = tid >> 5;
    const int lane = tid & 31;
    const int half = lane >> 4;
    const int l15  = lane & 15;

    const int  tiles_n = N >> 4;                       // multiple of 8
    const long mt      = ((long)blockIdx.x * 8) / tiles_n;   // block-uniform
    const int  nt      = (int)(((long)blockIdx.x * 8 + wave) % tiles_n);
    if (mt * 16 >= M) return;                          // block-uniform

    // stage A slab: 16*K*2 bytes contiguous (rows are consecutive in memory)
    {
        const char* gsrc = (const char*)(A + (size_t)mt * 16 * K);
        char* lbase = (char*)atile;
        constexpr int CHUNKS = 16 * K * 2 / 16;        // 16B chunks
#pragma unroll
        for (int i = 0; i < CHUNKS / 256; ++i) {
            int c = tid + i * 256;
            async_copy16(gsrc + (size_t)c * 16, lbase + (size_t)c * 16);
        }
        async_wait0();
        __syncthreads();
    }

    const int col = nt * 16 + l15;                     // output column (lane->N)
    v8f c;
    const float bv = bias[col];
#pragma unroll
    for (int r = 0; r < 8; ++r) c[r] = bv;

    const _Float16* brow = Bw + (size_t)col * K;       // lane -> N row of weights

#pragma unroll
    for (int kt = 0; kt < K; kt += 32) {
        AFrag a;   // K = (e>=8?16:0) + 8*half + (e&7): two 8-half runs from LDS
        const v8h* pa = (const v8h*)&atile[l15 * K + kt + half * 8];
        a.h[0] = pa[0];
        a.h[1] = pa[2];
        AFrag bf; bf.h[0] = *(const v8h*)(brow + kt + half * 8); bf.h[1] = *(const v8h*)(brow + kt + 16 + half * 8);
        const v16h b = bf.v;  // K=(e>=8?16:0)+8*half+(e&7)
        c = __builtin_amdgcn_wmma_f32_16x16x32_f16(false, a.v, false, b,
                                                   (short)0, c, false, false);
    }

    float* drow = D + (size_t)(mt * 16) * N + nt * 16 + l15;
#pragma unroll
    for (int r = 0; r < 8; ++r)
        drow[(size_t)(r + 8 * half) * N] = c[r];       // M = r + 8*half
}

// ---------------------------------------------------------------------------
// LSTM scan. Each block owns 16 batch rows and runs all T steps internally
// (recurrence is independent per batch row -> no grid sync).
// Wave w owns hidden units j in [w*JPW,(w+1)*JPW) and computes i/f/g/o tiles
// for that slice, so cell math is lane-aligned and c-state lives in VGPRs.
// The per-step gate slab gates_x[:, t, :] is pumped through LDS with
// async-to-LDS DMA overlapped with the WMMA recurrence of the current step.
// ---------------------------------------------------------------------------
template <int H, int NWAVES, bool WRITE_ALL>
__global__ __launch_bounds__(NWAVES * 32)
void lstm_scan_kernel(const float* __restrict__ gates_x,  // [B*T, 4H], row=b*T+t
                      const _Float16* __restrict__ Whh,   // [4H, H] fp16
                      _Float16* __restrict__ out_all,     // [B*T, H] fp16
                      float* __restrict__ out_last,       // [B, H] f32
                      int T) {
    constexpr int ROWS    = 16;
    constexpr int NG      = 4 * H;
    constexpr int JPW     = H / NWAVES;   // hidden units per wave
    constexpr int TPG     = JPW / 16;     // 16-col tiles per gate per wave
    constexpr int KSTEPS  = H / 32;
    constexpr int THREADS = NWAVES * 32;
    constexpr int CPR     = NG / 4;       // 16B chunks per gx row
    constexpr int PF_IT   = (ROWS * CPR) / THREADS;
    static_assert(TPG >= 1, "need >=16 hidden units per wave");
    static_assert((ROWS * CPR) % THREADS == 0, "prefetch mapping");

    __shared__ _Float16 hbuf[ROWS][H];
    __shared__ float    gxbuf[ROWS][NG];  // gate slab for current step

    const int tid  = threadIdx.x;
    const int wave = tid >> 5;
    const int lane = tid & 31;
    const int half = lane >> 4;
    const int l15  = lane & 15;
    const int b0   = blockIdx.x * ROWS;
    const int jbase = wave * JPW;

    for (int i = tid; i < ROWS * H; i += THREADS)
        ((_Float16*)hbuf)[i] = (_Float16)0.0f;

    // prefetch gate slab for t = 0
#pragma unroll
    for (int i = 0; i < PF_IT; ++i) {
        int c = tid + i * THREADS;
        int m = c / CPR, k = c % CPR;
        async_copy16(gates_x + ((size_t)(b0 + m) * T + 0) * NG + k * 4,
                     &gxbuf[m][k * 4]);
    }
    async_wait0();

    v8f cstate[TPG];
#pragma unroll
    for (int q = 0; q < TPG; ++q)
#pragma unroll
        for (int r = 0; r < 8; ++r) cstate[q][r] = 0.0f;

    __syncthreads();

    for (int t = 0; t < T; ++t) {
        // 1) consume gate slab (input contribution + biases) into accumulators
        v8f acc[4][TPG];
#pragma unroll
        for (int g = 0; g < 4; ++g)
#pragma unroll
            for (int q = 0; q < TPG; ++q) {
                const int col = g * H + jbase + q * 16 + l15;
                v8f cc;
#pragma unroll
                for (int r = 0; r < 8; ++r)
                    cc[r] = gxbuf[r + 8 * half][col];
                acc[g][q] = cc;
            }
        __syncthreads();  // slab consumed; safe to overwrite

        // 2) kick off async DMA of next step's slab, overlapped with WMMA
        if (t + 1 < T) {
#pragma unroll
            for (int i = 0; i < PF_IT; ++i) {
                int c = tid + i * THREADS;
                int m = c / CPR, k = c % CPR;
                async_copy16(gates_x + ((size_t)(b0 + m) * T + (t + 1)) * NG + k * 4,
                             &gxbuf[m][k * 4]);
            }
        }

        // 3) recurrent h_{t-1} @ Whh^T via WMMA (A from LDS, B from L2)
#pragma unroll
        for (int kt = 0; kt < KSTEPS; ++kt) {
            AFrag a;  // A row = batch row l15; K = (e>=8?16:0)+8*half+(e&7)
            const v8h* pa = (const v8h*)&hbuf[l15][kt * 32 + half * 8];
            a.h[0] = pa[0];
            a.h[1] = pa[2];
#pragma unroll
            for (int g = 0; g < 4; ++g)
#pragma unroll
                for (int q = 0; q < TPG; ++q) {
                    const int col = g * H + jbase + q * 16 + l15;
                    AFrag bf;
                    bf.h[0] = *(const v8h*)&Whh[(size_t)col * H + kt * 32 + half * 8];
                    bf.h[1] = *(const v8h*)&Whh[(size_t)col * H + kt * 32 + 16 + half * 8];
                    const v16h b = bf.v;
                    acc[g][q] = __builtin_amdgcn_wmma_f32_16x16x32_f16(
                        false, a.v, false, b, (short)0, acc[g][q], false, false);
                }
        }
        async_wait0();    // next slab landed (per-wave ASYNCcnt)
        __syncthreads();  // all waves done reading hbuf; gxbuf fully written

        // 4) lane-aligned LSTM cell math; c-state stays in registers
#pragma unroll
        for (int q = 0; q < TPG; ++q) {
#pragma unroll
            for (int r = 0; r < 8; ++r) {
                float ii = sigmoidf_(acc[0][q][r]);
                float ff = sigmoidf_(acc[1][q][r]);
                float gg = tanhf(acc[2][q][r]);
                float oo = sigmoidf_(acc[3][q][r]);
                float cc = ff * cstate[q][r] + ii * gg;
                cstate[q][r] = cc;
                float hh = oo * tanhf(cc);
                const int m = r + 8 * half;
                const int j = jbase + q * 16 + l15;
                hbuf[m][j] = (_Float16)hh;
                if (WRITE_ALL) {
                    out_all[((size_t)(b0 + m) * T + t) * H + j] = (_Float16)hh;
                } else if (t == T - 1) {
                    out_last[(size_t)(b0 + m) * H + j] = hh;
                }
            }
        }
        __syncthreads();  // hbuf now holds h_t
    }
}

// ---------------------------------------------------------------------------
// logits = h2 @ Wlin^T + blin ; softmax over C=5. One thread per batch row.
// ---------------------------------------------------------------------------
__global__ __launch_bounds__(256)
void head_softmax_kernel(const float* __restrict__ h2,
                         const float* __restrict__ Wlin,
                         const float* __restrict__ blin,
                         float* __restrict__ out, int B, int H2) {
    constexpr int C = 5;
    int b = blockIdx.x * blockDim.x + threadIdx.x;
    if (b >= B) return;
    const float* hr = h2 + (size_t)b * H2;
    float logits[C];
#pragma unroll
    for (int c = 0; c < C; ++c) {
        const float* wr = Wlin + (size_t)c * H2;
        float acc = blin[c];
        for (int k = 0; k < H2; ++k) acc += hr[k] * wr[k];
        logits[c] = acc;
    }
    float mx = logits[0];
#pragma unroll
    for (int c = 1; c < C; ++c) mx = fmaxf(mx, logits[c]);
    float sum = 0.0f;
#pragma unroll
    for (int c = 0; c < C; ++c) {
        logits[c] = __expf(logits[c] - mx);
        sum += logits[c];
    }
    float inv = 1.0f / sum;
#pragma unroll
    for (int c = 0; c < C; ++c) out[(size_t)b * C + c] = logits[c] * inv;
}

// ---------------------------------------------------------------------------
extern "C" void kernel_launch(void* const* d_in, const int* in_sizes, int n_in,
                              void* d_out, int out_size, void* d_ws, size_t ws_size,
                              hipStream_t stream) {
    (void)in_sizes; (void)n_in; (void)out_size; (void)ws_size;

    const int*   x    = (const int*)  d_in[0];
    const float* emb  = (const float*)d_in[1];
    const float* Wih1 = (const float*)d_in[2];
    const float* Whh1 = (const float*)d_in[3];
    const float* bih1 = (const float*)d_in[4];
    const float* bhh1 = (const float*)d_in[5];
    const float* Wih2 = (const float*)d_in[6];
    const float* Whh2 = (const float*)d_in[7];
    const float* bih2 = (const float*)d_in[8];
    const float* bhh2 = (const float*)d_in[9];
    const float* Wlin = (const float*)d_in[10];
    const float* blin = (const float*)d_in[11];
    float* out = (float*)d_out;

    constexpr int  B = 256, T = 512, E = 256, H1 = 256, H2 = 64;
    constexpr long BT = (long)B * T;

    char* ws = (char*)d_ws;
    size_t off = 0;
    auto take = [&](size_t bytes) -> char* {
        char* p = ws + off;
        off += (bytes + 255) & ~(size_t)255;
        return p;
    };
    _Float16* xe     = (_Float16*)take((size_t)BT * E * 2);
    _Float16* Wih1h  = (_Float16*)take((size_t)4 * H1 * E * 2);
    _Float16* Whh1h  = (_Float16*)take((size_t)4 * H1 * H1 * 2);
    _Float16* Wih2h  = (_Float16*)take((size_t)4 * H2 * H1 * 2);
    _Float16* Whh2h  = (_Float16*)take((size_t)4 * H2 * H2 * 2);
    float*    bsum1  = (float*)   take((size_t)4 * H1 * 4);
    float*    bsum2  = (float*)   take((size_t)4 * H2 * 4);
    float*    gates1 = (float*)   take((size_t)BT * 4 * H1 * 4);
    _Float16* out1h  = (_Float16*)take((size_t)BT * H1 * 2);
    float*    gates2 = (float*)   take((size_t)BT * 4 * H2 * 4);
    float*    h2last = (float*)   take((size_t)B * H2 * 4);

    // 1) weight conversions f32 -> f16 (tiny)
    {
        int n1 = 4 * H1 * E;
        int n2 = 4 * H1 * H1;
        int n3 = 4 * H2 * H1;
        int n4 = 4 * H2 * H2;
        cvt_f16_kernel<<<(n1 / 4 + 255) / 256, 256, 0, stream>>>(Wih1, Wih1h, n1);
        cvt_f16_kernel<<<(n2 / 4 + 255) / 256, 256, 0, stream>>>(Whh1, Whh1h, n2);
        cvt_f16_kernel<<<(n3 / 4 + 255) / 256, 256, 0, stream>>>(Wih2, Wih2h, n3);
        cvt_f16_kernel<<<(n4 / 4 + 255) / 256, 256, 0, stream>>>(Whh2, Whh2h, n4);
        bias_sum_kernel<<<(4 * H1 + 255) / 256, 256, 0, stream>>>(bih1, bhh1, bsum1, 4 * H1);
        bias_sum_kernel<<<(4 * H2 + 255) / 256, 256, 0, stream>>>(bih2, bhh2, bsum2, 4 * H2);
    }

    // 2) embedding gather -> fp16
    {
        long total = BT * E;
        long threads = total / 8;
        int blocks = (int)((threads + 255) / 256);
        embed_f16_kernel<<<blocks, 256, 0, stream>>>(x, emb, xe, BT, E);
    }

    // 3) gates1 = xe @ Wih1^T + (bih1+bhh1)   [BT, 1024]
    {
        long tiles = (BT / 16) * (4 * H1 / 16);
        gemm_bias_f16_kernel<E><<<(int)(tiles / 8), 256, 0, stream>>>(
            xe, Wih1h, bsum1, gates1, (int)BT, 4 * H1);
    }

    // 4) layer-1 scan: 16 blocks x 16 waves, writes out1 (fp16)
    lstm_scan_kernel<H1, 16, true><<<B / 16, 512, 0, stream>>>(
        gates1, Whh1h, out1h, nullptr, T);

    // 5) gates2 = out1 @ Wih2^T + (bih2+bhh2)  [BT, 256]
    {
        long tiles = (BT / 16) * (4 * H2 / 16);
        gemm_bias_f16_kernel<H1><<<(int)(tiles / 8), 256, 0, stream>>>(
            out1h, Wih2h, bsum2, gates2, (int)BT, 4 * H2);
    }

    // 6) layer-2 scan: only final hidden needed (f32)
    lstm_scan_kernel<H2, 4, false><<<B / 16, 128, 0, stream>>>(
        gates2, Whh2h, nullptr, h2last, T);

    // 7) logits + softmax
    head_softmax_kernel<<<(B + 255) / 256, 256, 0, stream>>>(
        h2last, Wlin, blin, out, B, H2);
}
